// MPNN_50414326120521
// MI455X (gfx1250) — compile-verified, matrix operand order checked by reading, not run on hardware
//
#include <hip/hip_runtime.h>
#include <hip/hip_bf16.h>

typedef __attribute__((ext_vector_type(16))) _Float16 v16h;
typedef __attribute__((ext_vector_type(8)))  _Float16 v8h;
typedef __attribute__((ext_vector_type(8)))  float    v8f;

#define DM      128
#define WAVES   8
#define EDGE_IN 384

// Dynamic LDS layouts (bytes). Order: f16 weights(T), f32 bias/gamma/beta, f16 act slabs.
static constexpr size_t LDS_EDGE =
    (size_t)(DM * EDGE_IN + 2 * DM * DM) * 2 + 5 * DM * 4 + (size_t)WAVES * 16 * EDGE_IN * 2;
static constexpr size_t LDS_NODE =
    (size_t)(3 * DM * DM) * 2 + 5 * DM * 4 + (size_t)WAVES * 16 * DM * 2;

// ---------------------------------------------------------------------------
// Intra-wave LDS producer->consumer sync: LDS ops are in-order per wave on
// CDNA5, so draining DScnt is sufficient (no workgroup barrier needed).
__device__ __forceinline__ void wave_lds_sync() {
    __builtin_amdgcn_wave_barrier();
    asm volatile("s_wait_dscnt 0x0" ::: "memory");
    __builtin_amdgcn_wave_barrier();
}

// 16-lane (half-wave) butterfly sum; xor masks 1..8 stay within the group.
__device__ __forceinline__ float red16(float v) {
    v += __shfl_xor(v, 1, 32);
    v += __shfl_xor(v, 2, 32);
    v += __shfl_xor(v, 4, 32);
    v += __shfl_xor(v, 8, 32);
    return v;
}

// Build one wave32 WMMA 16-bit fragment (A row / B column share the same
// K-striping): lane group h = lane>>4; element i maps to K = (i>>3)*16 + h*8 + (i&7).
// With row-major (A) / K-major-per-column (B = W^T) LDS storage this is two
// contiguous 16B ds_load_b128s.
__device__ __forceinline__ v16h load_frag(const _Float16* rowBase, int k0, int h) {
    const v8h lo = *(const v8h*)(rowBase + k0 + h * 8);
    const v8h hi = *(const v8h*)(rowBase + k0 + 16 + h * 8);
    return __builtin_shufflevector(lo, hi, 0, 1, 2, 3, 4, 5, 6, 7,
                                   8, 9, 10, 11, 12, 13, 14, 15);
}

// One wave computes a 16 x 128 tile: acc = act(16 x kdim) @ W. First K-step
// uses SRC2 = inline 0 (no accumulator-broadcast movs); bias is applied in the
// epilogues instead.
__device__ __forceinline__ void gemm16x128(const _Float16* act, int actStride,
                                           const _Float16* WT, int kdim,
                                           v8f acc[8], int lane) {
    const int h = lane >> 4, nlo = lane & 15;
    const _Float16* arow = act + nlo * actStride;  // lane holds row m = lane&15
    {
        const v16h af = load_frag(arow, 0, h);
#pragma unroll
        for (int t = 0; t < 8; ++t) {
            const v16h bf = load_frag(WT + (t * 16 + nlo) * kdim, 0, h);
            const v8f z = {};  // lowers to SRC2 = inline 0
            acc[t] = __builtin_amdgcn_wmma_f32_16x16x32_f16(
                false, af, false, bf, (short)0, z, false, false);
        }
    }
    for (int k0 = 32; k0 < kdim; k0 += 32) {
        const v16h af = load_frag(arow, k0, h);
#pragma unroll
        for (int t = 0; t < 8; ++t) {
            const v16h bf = load_frag(WT + (t * 16 + nlo) * kdim, k0, h);
            acc[t] = __builtin_amdgcn_wmma_f32_16x16x32_f16(
                false, af, false, bf, (short)0, acc[t], false, false);
        }
    }
}

// Epilogue: acc + bias, ReLU, f16 writeback to the activation slab.
// (C/D layout: lane group h, VGPR r -> row r+8h, lane&15 -> col.)
__device__ __forceinline__ void store_act_bias_relu(const v8f acc[8], const float* bias,
                                                    _Float16* slab, int stride, int lane) {
    const int h = lane >> 4, nlo = lane & 15;
#pragma unroll
    for (int t = 0; t < 8; ++t) {
        const float bv = bias[nlo + 16 * t];
#pragma unroll
        for (int r = 0; r < 8; ++r)
            slab[(r + 8 * h) * stride + (nlo + 16 * t)] =
                (_Float16)fmaxf(acc[t][r] + bv, 0.0f);
    }
}

// Bias must land BEFORE the LayerNorm statistics (LN is not invariant to a
// per-column shift). Then normalize with gamma/beta, in-register.
__device__ __forceinline__ void bias_layernorm_acc(v8f acc[8], const float* bias,
                                                   const float* g, const float* beta,
                                                   int lane) {
    const int nlo = lane & 15;
    float s[8], q[8];
#pragma unroll
    for (int r = 0; r < 8; ++r) { s[r] = 0.0f; q[r] = 0.0f; }
#pragma unroll
    for (int t = 0; t < 8; ++t) {
        const float bv = bias[nlo + 16 * t];
#pragma unroll
        for (int r = 0; r < 8; ++r) {
            const float v = acc[t][r] + bv;
            acc[t][r] = v;
            s[r] += v;
            q[r] += v * v;
        }
    }
#pragma unroll
    for (int r = 0; r < 8; ++r) {
        const float sum = red16(s[r]);
        const float ssq = red16(q[r]);
        const float mu = sum * (1.0f / 128.0f);
        const float var = ssq * (1.0f / 128.0f) - mu * mu;
        s[r] = mu;
        q[r] = rsqrtf(var + 1e-5f);
    }
#pragma unroll
    for (int t = 0; t < 8; ++t) {
        const float gv = g[nlo + 16 * t];
        const float bv = beta[nlo + 16 * t];
#pragma unroll
        for (int r = 0; r < 8; ++r)
            acc[t][r] = (acc[t][r] - s[r]) * q[r] * gv + bv;
    }
}

// Convert 8 consecutive floats (two float4 loads) to one v8h.
__device__ __forceinline__ v8h cvt8(const float4 a, const float4 b) {
    v8h p;
    p[0] = (_Float16)a.x; p[1] = (_Float16)a.y; p[2] = (_Float16)a.z; p[3] = (_Float16)a.w;
    p[4] = (_Float16)b.x; p[5] = (_Float16)b.y; p[6] = (_Float16)b.z; p[7] = (_Float16)b.w;
    return p;
}

// ---------------------------------------------------------------------------
// Edge stage: gather -> 384->128->128->128 MLP -> LayerNorm -> fused
// reshape(N,128,16).sum(-1): node_in.flat[row*8 + c] = 16-wide chunk sums.
__global__ __launch_bounds__(256, 1) void mpnn_edge_kernel(
    const float* __restrict__ x, const float* __restrict__ ea,
    const int* __restrict__ recv, const int* __restrict__ send,
    const float* __restrict__ W0, const float* __restrict__ b0,
    const float* __restrict__ W1, const float* __restrict__ b1,
    const float* __restrict__ W2, const float* __restrict__ b2,
    const float* __restrict__ g, const float* __restrict__ beta,
    float* __restrict__ node_in, int E) {
    extern __shared__ _Float16 smem[];
    _Float16* sW0T = smem;                         // [128][384]
    _Float16* sW1T = sW0T + DM * EDGE_IN;          // [128][128]
    _Float16* sW2T = sW1T + DM * DM;               // [128][128]
    float* fb = (float*)(sW2T + DM * DM);
    float* sb0 = fb;        float* sb1 = sb0 + DM; float* sb2 = sb1 + DM;
    float* sg = sb2 + DM;   float* sbt = sg + DM;
    _Float16* slabs = (_Float16*)(sbt + DM);       // WAVES * [16][384]

    const int tid = threadIdx.x;
    for (int i = tid; i < DM * EDGE_IN; i += blockDim.x) {  // W0^T
        const int n = i / EDGE_IN, k = i % EDGE_IN;
        sW0T[i] = (_Float16)W0[k * DM + n];
    }
    for (int i = tid; i < DM * DM; i += blockDim.x) {       // W1^T, W2^T
        const int n = i >> 7, k = i & 127;
        sW1T[i] = (_Float16)W1[k * DM + n];
        sW2T[i] = (_Float16)W2[k * DM + n];
    }
    for (int i = tid; i < DM; i += blockDim.x) {
        sb0[i] = b0[i]; sb1[i] = b1[i]; sb2[i] = b2[i];
        sg[i] = g[i];   sbt[i] = beta[i];
    }
    __syncthreads();

    const int wave = tid >> 5, lane = tid & 31;
    const int h = lane >> 4, nlo = lane & 15;
    _Float16* slab = slabs + wave * 16 * EDGE_IN;
    const int numTiles = (E + 15) >> 4;
    const int tileStep = gridDim.x * WAVES;

    for (int tile = blockIdx.x * WAVES + wave; tile < numTiles; tile += tileStep) {
        const int e0 = tile << 4;

        // Warm L2 for the next tile's edge_attr rows (the only real HBM
        // stream): 16 rows * 512 B = 8 KB = 64 cachelines, 2 per lane.
        const int tnext = tile + tileStep;
        if (tnext < numTiles) {
            const char* p = (const char*)(ea + (size_t)(tnext << 4) * DM);
            __builtin_prefetch(p + lane * 256, 0, 1);
            __builtin_prefetch(p + lane * 256 + 128, 0, 1);
        }

        // Stage [16][384] f16 inputs: each lane owns 8-column chunks ->
        // two global b128 loads + one ds_store_b128 per chunk, coalesced.
        for (int c = lane; c < 16 * (EDGE_IN / 8); c += 32) {
            const int row = c / (EDGE_IN / 8);
            const int cb = (c % (EDGE_IN / 8)) * 8;
            int e = e0 + row;
            if (e >= E) e = E - 1;
            const float* src;
            int off;
            if (cb < 128)      { src = x + (size_t)recv[e] * DM;  off = cb; }
            else if (cb < 256) { src = x + (size_t)send[e] * DM;  off = cb - 128; }
            else               { src = ea + (size_t)e * DM;       off = cb - 256; }
            const float4 a = ((const float4*)(src + off))[0];
            const float4 b = ((const float4*)(src + off))[1];
            *(v8h*)(slab + row * EDGE_IN + cb) = cvt8(a, b);
        }
        wave_lds_sync();

        v8f acc[8];
        gemm16x128(slab, EDGE_IN, sW0T, EDGE_IN, acc, lane);   // layer 0
        store_act_bias_relu(acc, sb0, slab, EDGE_IN, lane);
        wave_lds_sync();
        gemm16x128(slab, EDGE_IN, sW1T, DM, acc, lane);        // layer 1
        store_act_bias_relu(acc, sb1, slab, EDGE_IN, lane);
        wave_lds_sync();
        gemm16x128(slab, EDGE_IN, sW2T, DM, acc, lane);        // layer 2
        bias_layernorm_acc(acc, sb2, sg, sbt, lane);

        // Fused reshape-sum: chunk c = tile t; reduce across the 16-lane group.
#pragma unroll
        for (int t = 0; t < 8; ++t) {
#pragma unroll
            for (int r = 0; r < 8; ++r) {
                const float c = red16(acc[t][r]);
                const int erow = e0 + r + 8 * h;
                if (nlo == t && erow < E)
                    node_in[(size_t)erow * 8 + t] = c;
            }
        }
    }
}

// ---------------------------------------------------------------------------
// Node stage: [N,128] -> 3x(128x128) MLP -> LayerNorm -> d_out.
__global__ __launch_bounds__(256, 1) void mpnn_node_kernel(
    const float* __restrict__ node_in,
    const float* __restrict__ W0, const float* __restrict__ b0,
    const float* __restrict__ W1, const float* __restrict__ b1,
    const float* __restrict__ W2, const float* __restrict__ b2,
    const float* __restrict__ g, const float* __restrict__ beta,
    float* __restrict__ out, int N) {
    extern __shared__ _Float16 smem[];
    _Float16* sW0T = smem;
    _Float16* sW1T = sW0T + DM * DM;
    _Float16* sW2T = sW1T + DM * DM;
    float* fb = (float*)(sW2T + DM * DM);
    float* sb0 = fb;        float* sb1 = sb0 + DM; float* sb2 = sb1 + DM;
    float* sg = sb2 + DM;   float* sbt = sg + DM;
    _Float16* slabs = (_Float16*)(sbt + DM);  // WAVES * [16][128]

    const int tid = threadIdx.x;
    for (int i = tid; i < DM * DM; i += blockDim.x) {
        const int n = i >> 7, k = i & 127;
        sW0T[i] = (_Float16)W0[k * DM + n];
        sW1T[i] = (_Float16)W1[k * DM + n];
        sW2T[i] = (_Float16)W2[k * DM + n];
    }
    for (int i = tid; i < DM; i += blockDim.x) {
        sb0[i] = b0[i]; sb1[i] = b1[i]; sb2[i] = b2[i];
        sg[i] = g[i];   sbt[i] = beta[i];
    }
    __syncthreads();

    const int wave = tid >> 5, lane = tid & 31;
    const int h = lane >> 4, nlo = lane & 15;
    _Float16* slab = slabs + wave * 16 * DM;
    const int numTiles = (N + 15) >> 4;

    for (int tile = blockIdx.x * WAVES + wave; tile < numTiles; tile += gridDim.x * WAVES) {
        const int n0 = tile << 4;
        // Stage [16][128] f16: 8-column chunks, b128 loads + b128 LDS stores.
        for (int c = lane; c < 16 * (DM / 8); c += 32) {
            const int row = c >> 4;
            const int cb = (c & 15) * 8;
            int n = n0 + row;
            if (n >= N) n = N - 1;
            const float* src = node_in + (size_t)n * DM + cb;
            const float4 a = ((const float4*)src)[0];
            const float4 b = ((const float4*)src)[1];
            *(v8h*)(slab + row * DM + cb) = cvt8(a, b);
        }
        wave_lds_sync();

        v8f acc[8];
        gemm16x128(slab, DM, sW0T, DM, acc, lane);
        store_act_bias_relu(acc, sb0, slab, DM, lane);
        wave_lds_sync();
        gemm16x128(slab, DM, sW1T, DM, acc, lane);
        store_act_bias_relu(acc, sb1, slab, DM, lane);
        wave_lds_sync();
        gemm16x128(slab, DM, sW2T, DM, acc, lane);
        bias_layernorm_acc(acc, sb2, sg, sbt, lane);

#pragma unroll
        for (int t = 0; t < 8; ++t)
#pragma unroll
            for (int r = 0; r < 8; ++r) {
                const int nrow = n0 + r + 8 * h;
                if (nrow < N)
                    out[(size_t)nrow * DM + (nlo + 16 * t)] = acc[t][r];
            }
    }
}

// ---------------------------------------------------------------------------
extern "C" void kernel_launch(void* const* d_in, const int* in_sizes, int n_in,
                              void* d_out, int out_size, void* d_ws, size_t ws_size,
                              hipStream_t stream) {
    const float* x    = (const float*)d_in[0];
    const float* ea   = (const float*)d_in[1];
    const int*   recv = (const int*)d_in[2];
    const int*   send = (const int*)d_in[3];
    // d_in[4] = n_atoms scalar (unused; derive from sizes)
    const float* eW0 = (const float*)d_in[5];
    const float* eb0 = (const float*)d_in[6];
    const float* eW1 = (const float*)d_in[7];
    const float* eb1 = (const float*)d_in[8];
    const float* eW2 = (const float*)d_in[9];
    const float* eb2 = (const float*)d_in[10];
    const float* eg  = (const float*)d_in[11];
    const float* ebt = (const float*)d_in[12];
    const float* nW0 = (const float*)d_in[13];
    const float* nb0 = (const float*)d_in[14];
    const float* nW1 = (const float*)d_in[15];
    const float* nb1 = (const float*)d_in[16];
    const float* nW2 = (const float*)d_in[17];
    const float* nb2 = (const float*)d_in[18];
    const float* ng  = (const float*)d_in[19];
    const float* nbt = (const float*)d_in[20];

    const int N = in_sizes[0] / DM;
    const int E = in_sizes[1] / DM;

    float* node_in = (float*)d_ws;  // [N][128] fused messages
    float* out = (float*)d_out;

    mpnn_edge_kernel<<<256, 256, LDS_EDGE, stream>>>(
        x, ea, recv, send, eW0, eb0, eW1, eb1, eW2, eb2, eg, ebt, node_in, E);
    mpnn_node_kernel<<<80, 256, LDS_NODE, stream>>>(
        node_in, nW0, nb0, nW1, nb1, nW2, nb2, ng, nbt, out, N);
}
